// CrossAttention_1288490188990
// MI455X (gfx1250) — matrix operand order repaired and checked by reading, not compiled, not run on hardware
//
#include <hip/hip_runtime.h>

// ---------------------------------------------------------------------------
// Types for CDNA5 WMMA (wave32): v_wmma_f32_16x16x32_bf16
// ---------------------------------------------------------------------------
typedef __attribute__((ext_vector_type(16))) __bf16 v16bf;
typedef __attribute__((ext_vector_type(8)))  __bf16 v8bf;
typedef __attribute__((ext_vector_type(8)))  float  v8f;

static __device__ __forceinline__ __bf16 f2bf(float f) { return (__bf16)f; }
static __device__ __forceinline__ float  bf2f(__bf16 b) { return (float)b; }

static __device__ __forceinline__ v8f v8f_zero() {
    v8f z = {0.f, 0.f, 0.f, 0.f, 0.f, 0.f, 0.f, 0.f};
    return z;
}

// CDNA5 async copy global -> LDS (ASYNCcnt tracked), 16 bytes per lane.
static __device__ __forceinline__ void async_copy_b128(unsigned lds_off, const void* gptr) {
    unsigned long long ga = (unsigned long long)(uintptr_t)gptr;
    asm volatile("global_load_async_to_lds_b128 %0, %1, off"
                 :: "v"(lds_off), "v"(ga) : "memory");
}
static __device__ __forceinline__ void wait_async0() {
    asm volatile("s_wait_asynccnt 0x0" ::: "memory");
}
static __device__ __forceinline__ unsigned lds_addr32(const void* p) {
    return (unsigned)(uintptr_t)p;   // generic LDS pointer: low 32 bits = LDS offset
}

// A-operand 16x32 bf16 tile (row-major source, leading dim lda).
// ISA layout: lane<16 -> row=lane, K {0..7,16..23}; lane>=16 -> row=lane-16, K {8..15,24..31}.
static __device__ __forceinline__ v16bf loadA_bf16(const __bf16* base, int lda) {
    int lane = threadIdx.x & 31;
    int row  = lane & 15;
    int k0   = (lane >> 4) << 3;              // 0 or 8
    const __bf16* p = base + (size_t)row * lda + k0;
    v8bf lo = *reinterpret_cast<const v8bf*>(p);
    v8bf hi = *reinterpret_cast<const v8bf*>(p + 16);
    v16bf a;
#pragma unroll
    for (int i = 0; i < 8; ++i) { a[i] = lo[i]; a[i + 8] = hi[i]; }
    return a;
}

// B-operand 32x16 bf16 tile where the source holds B^T row-major (row n = 32 contiguous K).
// ISA layout: lane&15 = column n; lanes<16 hold K {0..7,16..23}, lanes>=16 hold K {8..15,24..31}.
static __device__ __forceinline__ v16bf loadB_bf16(const __bf16* baseT, int ldb) {
    int lane = threadIdx.x & 31;
    int n    = lane & 15;
    int kb   = (lane >> 4) << 3;              // 0 or 8
    const __bf16* p = baseT + (size_t)n * ldb + kb;
    v8bf lo = *reinterpret_cast<const v8bf*>(p);
    v8bf hi = *reinterpret_cast<const v8bf*>(p + 16);
    v16bf b;
#pragma unroll
    for (int i = 0; i < 8; ++i) { b[i] = lo[i]; b[i + 8] = hi[i]; }
    return b;
}

static __device__ __forceinline__ v8f wmma_bf16(v16bf a, v16bf b, v8f c) {
    return __builtin_amdgcn_wmma_f32_16x16x32_bf16(false, a, false, b, (short)0, c,
                                                   false, false);
}

// ---------------------------------------------------------------------------
// Kernel: fp32 -> bf16 bulk convert (8 elements / thread)
// ---------------------------------------------------------------------------
__global__ __launch_bounds__(256) void f32_to_bf16_kernel(
    const float* __restrict__ x, __bf16* __restrict__ y, int n8) {
    int i = blockIdx.x * 256 + threadIdx.x;
    if (i < n8) {
        size_t base = (size_t)i * 8;
        v8bf o;
#pragma unroll
        for (int j = 0; j < 8; ++j) o[j] = f2bf(x[base + j]);
        *reinterpret_cast<v8bf*>(y + base) = o;
    }
}

// ---------------------------------------------------------------------------
// Kernel: fp32 [K x N] -> bf16 transposed [N x K]
// ---------------------------------------------------------------------------
__global__ void transpose_w_kernel(const float* __restrict__ W, __bf16* __restrict__ Wt,
                                   int K, int Ncols, int total) {
    int idx = blockIdx.x * 256 + threadIdx.x;
    if (idx < total) {
        int kk = idx / Ncols;
        int n  = idx % Ncols;
        Wt[(size_t)n * K + kk] = f2bf(W[idx]);
    }
}

// ---------------------------------------------------------------------------
// Generic 256x64 tiled GEMM:  C = A(bf16,[rows x K]) @ Bt(bf16,[cols x K])^T
// Block = 256 threads (8 waves); wave tile = 32 rows x 64 cols (2 row subtiles
// sharing each B fragment -> 8 WMMAs per 12 ds_load_b128 per k-step).
// Async-LDS double-buffered staging of A[256x32] and Bt[64x32] per k-step.
// MODE 0: dstF fp32 row-major (scores)
// MODE 1: dstH bf16 row-major (P@V)
// MODE 2: dstH bf16 row-major + bias (q/k projection)
// MODE 3: dstH bf16 + bias, stored as vt[b][d][m] with 2048 rows/batch (v proj)
// ---------------------------------------------------------------------------
template <int MODE>
__global__ __launch_bounds__(256) void gemm256x64_kernel(
    const __bf16* __restrict__ A, int lda,
    const __bf16* __restrict__ Bt, int ldb, int K,
    const float* __restrict__ bias,
    float* __restrict__ dstF, __bf16* __restrict__ dstH, int ldd) {
    __shared__ __bf16 sA[2][256 * 32];   // 2 x 16 KB
    __shared__ __bf16 sB[2][64 * 32];    // 2 x  4 KB

    const int t      = threadIdx.x;
    const int w      = t >> 5;
    const int rowblk = blockIdx.x * 256;
    const int colblk = blockIdx.y * 64;

    // Stage one 32-wide k-slice of A (256x32) and Bt (64x32) into LDS buffer `buf`.
    auto stage = [&](int buf, int kk) {
        // A: 1024 16-byte chunks, 4 per thread
#pragma unroll
        for (int i = 0; i < 4; ++i) {
            int chunk = t + 256 * i;
            int row = chunk >> 2, cp = chunk & 3;
            async_copy_b128(lds_addr32(&sA[buf][row * 32 + cp * 8]),
                            A + (size_t)(rowblk + row) * lda + kk + cp * 8);
        }
        // Bt: 256 16-byte chunks, 1 per thread
        {
            int row = t >> 2, cp = t & 3;
            async_copy_b128(lds_addr32(&sB[buf][row * 32 + cp * 8]),
                            Bt + (size_t)(colblk + row) * ldb + kk + cp * 8);
        }
    };

    v8f acc[2][4];
#pragma unroll
    for (int s = 0; s < 2; ++s)
#pragma unroll
        for (int j = 0; j < 4; ++j) acc[s][j] = v8f_zero();

    stage(0, 0);
    wait_async0();
    __syncthreads();

    for (int kk = 0; kk < K; kk += 32) {
        int buf = (kk >> 5) & 1;
        if (kk + 32 < K) stage(buf ^ 1, kk + 32);   // prefetch next slice (async)

        v16bf a0 = loadA_bf16(&sA[buf][(w * 32) * 32], 32);
        v16bf a1 = loadA_bf16(&sA[buf][(w * 32 + 16) * 32], 32);
#pragma unroll
        for (int j = 0; j < 4; ++j) {
            v16bf b = loadB_bf16(&sB[buf][(16 * j) * 32], 32);
            acc[0][j] = wmma_bf16(a0, b, acc[0][j]);
            acc[1][j] = wmma_bf16(a1, b, acc[1][j]);
        }

        wait_async0();
        __syncthreads();
    }

    const int lane = t & 31;
    const int n    = lane & 15;
    const int mo   = (lane >> 4) * 8;
#pragma unroll
    for (int s = 0; s < 2; ++s) {
        const int rowbase = rowblk + w * 32 + s * 16;
#pragma unroll
        for (int j = 0; j < 4; ++j) {
            int col = colblk + 16 * j + n;
            float bv = (MODE >= 2) ? bias[col] : 0.0f;
#pragma unroll
            for (int r = 0; r < 8; ++r) {
                int m = rowbase + mo + r;
                float val = acc[s][j][r] + bv;
                if (MODE == 0) {
                    dstF[(size_t)m * ldd + col] = val;
                } else if (MODE == 1 || MODE == 2) {
                    dstH[(size_t)m * ldd + col] = f2bf(val);
                } else {                   // MODE 3: vt[b][d][m], 2048 rows per batch
                    int bb = m >> 11, mm = m & 2047;
                    dstH[((size_t)bb * 512 + col) * 2048 + mm] = f2bf(val);
                }
            }
        }
    }
}

// ---------------------------------------------------------------------------
// Kernel: row softmax over 2048 cols; S fp32 -> P bf16. One block (256 thr) per row.
// ---------------------------------------------------------------------------
__global__ __launch_bounds__(256) void softmax_kernel(const float* __restrict__ S,
                                                      __bf16* __restrict__ P) {
    __shared__ float red[256];
    int t = threadIdx.x;
    const float* s = S + (size_t)blockIdx.x * 2048;
    __bf16*      p = P + (size_t)blockIdx.x * 2048;

    float v[8];
#pragma unroll
    for (int i = 0; i < 8; ++i) v[i] = s[t + 256 * i];

    float mx = v[0];
#pragma unroll
    for (int i = 1; i < 8; ++i) mx = fmaxf(mx, v[i]);
    red[t] = mx;
    __syncthreads();
    for (int off = 128; off > 0; off >>= 1) {
        if (t < off) red[t] = fmaxf(red[t], red[t + off]);
        __syncthreads();
    }
    mx = red[0];
    __syncthreads();

    float sum = 0.f;
#pragma unroll
    for (int i = 0; i < 8; ++i) { v[i] = __expf(v[i] - mx); sum += v[i]; }
    red[t] = sum;
    __syncthreads();
    for (int off = 128; off > 0; off >>= 1) {
        if (t < off) red[t] += red[t + off];
        __syncthreads();
    }
    float inv = 1.0f / red[0];
#pragma unroll
    for (int i = 0; i < 8; ++i) p[t + 256 * i] = f2bf(v[i] * inv);
}

// ---------------------------------------------------------------------------
// Kernel: LayerNorm (fp32 stats) + out projection (WMMA, Wo^T bf16) + bias.
// Block = 256 threads handles 32 rows; normalized rows staged in LDS.
// ---------------------------------------------------------------------------
__global__ __launch_bounds__(256) void lnproj_kernel(
    const __bf16* __restrict__ fused, const float* __restrict__ gamma,
    const float* __restrict__ beta, const __bf16* __restrict__ Wot,
    const float* __restrict__ bo, float* __restrict__ out) {
    __shared__ __bf16 norm[32 * 512];   // 32 KB
    __shared__ float  redA[256];
    __shared__ float  redB[256];

    int    t    = threadIdx.x;
    size_t row0 = (size_t)blockIdx.x * 32;
    int    r    = t >> 3;      // local row 0..31
    int    c    = t & 7;       // 64-element chunk within the row

    const __bf16* src = fused + (row0 + r) * 512 + c * 64;
    float s = 0.f, ss = 0.f;
    for (int i = 0; i < 64; ++i) {
        float x = bf2f(src[i]);
        s += x; ss += x * x;
    }
    redA[t] = s; redB[t] = ss;
    __syncthreads();
#pragma unroll
    for (int off = 4; off > 0; off >>= 1) {
        if (c < off) { redA[t] += redA[t + off]; redB[t] += redB[t + off]; }
        __syncthreads();
    }
    float mean = redA[r * 8] * (1.0f / 512.0f);
    float var  = redB[r * 8] * (1.0f / 512.0f) - mean * mean;
    float rs   = rsqrtf(var + 1e-5f);

    for (int i = 0; i < 64; ++i) {
        int col = c * 64 + i;
        float x = bf2f(src[i]);
        norm[r * 512 + col] = f2bf((x - mean) * rs * gamma[col] + beta[col]);
    }
    __syncthreads();

    // 8 waves: wave w -> row tile rt = w>>2 (0/1), col tiles at cb, cb+16 (cb = (w&3)*32)
    int w  = t >> 5;
    int rt = w >> 2;
    int cb = (w & 3) * 32;
    v8f acc[2];
    acc[0] = v8f_zero(); acc[1] = v8f_zero();

    for (int kk = 0; kk < 512; kk += 32) {
        v16bf a = loadA_bf16(&norm[rt * 16 * 512 + kk], 512);
#pragma unroll
        for (int j = 0; j < 2; ++j) {
            v16bf b = loadB_bf16(Wot + (size_t)(cb + 16 * j) * 512 + kk, 512);
            acc[j] = wmma_bf16(a, b, acc[j]);
        }
    }

    int lane = t & 31;
    int n    = lane & 15;
    int mo   = (lane >> 4) * 8;
#pragma unroll
    for (int j = 0; j < 2; ++j) {
        int col = cb + 16 * j + n;
        float bb = bo[col];
#pragma unroll
        for (int r8 = 0; r8 < 8; ++r8) {
            size_t m = row0 + rt * 16 + mo + r8;
            out[m * 128 + col] = acc[j][r8] + bb;
        }
    }
}

// ---------------------------------------------------------------------------
// Host: launch pipeline (all on `stream`; per-batch serialization bounds scratch).
// ---------------------------------------------------------------------------
extern "C" void kernel_launch(void* const* d_in, const int* in_sizes, int n_in,
                              void* d_out, int out_size, void* d_ws, size_t ws_size,
                              hipStream_t stream) {
    (void)in_sizes; (void)n_in; (void)out_size; (void)ws_size;

    const float* input1 = (const float*)d_in[0];
    const float* input2 = (const float*)d_in[1];
    const float* Wq     = (const float*)d_in[2];
    const float* bq     = (const float*)d_in[3];
    const float* Wk     = (const float*)d_in[4];
    const float* bk     = (const float*)d_in[5];
    const float* Wv     = (const float*)d_in[6];
    const float* bv     = (const float*)d_in[7];
    const float* gamma  = (const float*)d_in[8];
    const float* beta   = (const float*)d_in[9];
    const float* Wo     = (const float*)d_in[10];
    const float* bo     = (const float*)d_in[11];
    float*       out    = (float*)d_out;

    const int    Bb = 8, Nn = 2048, Mm = 2048, DIN = 512, DOUT = 512, DF = 128;
    const size_t BN = (size_t)Bb * Nn;        // 16384 rows

    char*  ws  = (char*)d_ws;
    size_t off = 0;
    auto take = [&](size_t bytes) -> char* {
        char* p = ws + off;
        off += (bytes + 255) & ~(size_t)255;
        return p;
    };
    __bf16* x1b    = (__bf16*)take(BN * DIN * 2);             // 16 MB
    __bf16* x2b    = (__bf16*)take(BN * DIN * 2);             // 16 MB
    __bf16* qb     = (__bf16*)take(BN * DOUT * 2);            // 16 MB
    __bf16* kb     = (__bf16*)take(BN * DOUT * 2);            // 16 MB
    __bf16* vt     = (__bf16*)take(BN * DOUT * 2);            // 16 MB, [b][d][m]
    __bf16* Wqt    = (__bf16*)take((size_t)DIN * DOUT * 2);
    __bf16* Wkt    = (__bf16*)take((size_t)DIN * DOUT * 2);
    __bf16* Wvt    = (__bf16*)take((size_t)DIN * DOUT * 2);
    __bf16* Wot    = (__bf16*)take((size_t)DOUT * DF * 2);
    float*  S      = (float*)take((size_t)Nn * Mm * 4);       // 16 MB (one batch)
    __bf16* P      = (__bf16*)take((size_t)Nn * Mm * 2);      //  8 MB (one batch)
    __bf16* fusedb = (__bf16*)take(BN * DOUT * 2);            // 16 MB

    // 1) convert activations, convert+transpose weights to bf16
    {
        int n8 = (int)(BN * DIN / 8);
        f32_to_bf16_kernel<<<(n8 + 255) / 256, 256, 0, stream>>>(input1, x1b, n8);
        f32_to_bf16_kernel<<<(n8 + 255) / 256, 256, 0, stream>>>(input2, x2b, n8);
    }
    transpose_w_kernel<<<(DIN * DOUT + 255) / 256, 256, 0, stream>>>(Wq, Wqt, DIN, DOUT, DIN * DOUT);
    transpose_w_kernel<<<(DIN * DOUT + 255) / 256, 256, 0, stream>>>(Wk, Wkt, DIN, DOUT, DIN * DOUT);
    transpose_w_kernel<<<(DIN * DOUT + 255) / 256, 256, 0, stream>>>(Wv, Wvt, DIN, DOUT, DIN * DOUT);
    transpose_w_kernel<<<(DOUT * DF + 255) / 256, 256, 0, stream>>>(Wo, Wot, DOUT, DF, DOUT * DF);

    // 2) q/k/v projections (WMMA, async-LDS pipelined)
    dim3 pg((unsigned)(BN / 256), DOUT / 64);
    gemm256x64_kernel<2><<<pg, 256, 0, stream>>>(x1b, DIN, Wqt, DIN, DIN, bq, nullptr, qb, DOUT);
    gemm256x64_kernel<2><<<pg, 256, 0, stream>>>(x2b, DIN, Wkt, DIN, DIN, bk, nullptr, kb, DOUT);
    gemm256x64_kernel<3><<<pg, 256, 0, stream>>>(x2b, DIN, Wvt, DIN, DIN, bv, nullptr, vt, DOUT);

    // 3) attention, one batch at a time (stream order serializes S/P reuse)
    for (int b = 0; b < Bb; ++b) {
        const __bf16* qbb = qb + (size_t)b * Nn * DOUT;
        const __bf16* kbb = kb + (size_t)b * Mm * DOUT;
        const __bf16* vtb = vt + (size_t)b * DOUT * Mm;
        __bf16*       fb  = fusedb + (size_t)b * Nn * DOUT;
        gemm256x64_kernel<0><<<dim3(Nn / 256, Mm / 64), 256, 0, stream>>>(
            qbb, DOUT, kbb, DOUT, DOUT, nullptr, S, nullptr, Mm);
        softmax_kernel<<<Nn, 256, 0, stream>>>(S, P);
        gemm256x64_kernel<1><<<dim3(Nn / 256, DOUT / 64), 256, 0, stream>>>(
            P, Mm, vtb, Mm, Mm, nullptr, nullptr, fb, DOUT);
    }

    // 4) LayerNorm + output projection (WMMA) + bias
    lnproj_kernel<<<(unsigned)(BN / 32), 256, 0, stream>>>(fusedb, gamma, beta, Wot, bo, out);
}
